// ConfidenceGuidedPropagation_88776974008406
// MI455X (gfx1250) — compile-verified, matrix operand order checked by reading, not run on hardware
//
#include <hip/hip_runtime.h>
#include <hip/hip_bf16.h>

#define D      128
#define NN     10000
#define NE     640000
#define NPAD   10112          // 158 * 64
#define TILE_M 64
#define LSTR   136            // LDS row stride (bf16 elems): 68 dwords -> conflict-free b128 frag reads

typedef __attribute__((ext_vector_type(16))) __bf16 v16bf;
typedef __attribute__((ext_vector_type(8)))  float  v8f;

union Frag { uint4 q[2]; v16bf v; };

static __device__ __forceinline__ unsigned short f2bf(float f) {
    union { float f; unsigned u; } a; a.f = f;
    unsigned u = a.u;
    u += 0x7FFFu + ((u >> 16) & 1u);          // round-to-nearest-even
    return (unsigned short)(u >> 16);
}

// ---------------- edge weights: exp(-|c_src - c_dst|) ----------------
__global__ __launch_bounds__(256) void edgew_kernel(const int* __restrict__ src,
                                                    const int* __restrict__ dst,
                                                    const float* __restrict__ conf,
                                                    float* __restrict__ ew) {
    int e = blockIdx.x * 256 + threadIdx.x;
    if (e < NE) ew[e] = __expf(-fabsf(conf[src[e]] - conf[dst[e]]));
}

// ---------------- f32 -> bf16 elementwise ----------------
__global__ __launch_bounds__(256) void cvt_kernel(const float* __restrict__ in,
                                                  unsigned short* __restrict__ out, int n) {
    int i = blockIdx.x * 256 + threadIdx.x;
    if (i < n) out[i] = f2bf(in[i]);
}

// ---------------- weight convert + transpose: W[k][n] f32 -> Wt[n][k] bf16 ----------------
__global__ __launch_bounds__(256) void wcvt_kernel(const float* __restrict__ W,
                                                   unsigned short* __restrict__ Wt) {
    int i = blockIdx.x * 256 + threadIdx.x;   // 16384
    int n = i >> 7, k = i & 127;
    Wt[n * D + k] = f2bf(W[k * D + n]);
}

// ---------------- fused 2-layer MLP: out = (relu(A@W1+b1))@W2 + b2 ----------------
__global__ __launch_bounds__(256)
void mlp2_kernel(const unsigned short* __restrict__ Abf,   // [NPAD][D] bf16
                 const unsigned short* __restrict__ W1t,   // [D][D] bf16, out-major
                 const float* __restrict__ b1,
                 const unsigned short* __restrict__ W2t,
                 const float* __restrict__ b2,
                 float* __restrict__ out)                   // [NPAD][D] f32
{
    __shared__ unsigned short lA[TILE_M * LSTR];
    __shared__ unsigned short lW[D * LSTR];

    const int t = threadIdx.x;
    const int rowBase = blockIdx.x * TILE_M;

    // A tile: 64x128 bf16 = 1024 x 16B chunks
    #pragma unroll
    for (int c = 0; c < 4; ++c) {
        int q = t + c * 256, r = q >> 4, k8 = (q & 15) << 3;
        *(uint4*)&lA[r * LSTR + k8] = *(const uint4*)&Abf[(size_t)(rowBase + r) * D + k8];
    }
    // W1t: 128x128 bf16 = 2048 chunks
    #pragma unroll
    for (int c = 0; c < 8; ++c) {
        int q = t + c * 256, r = q >> 4, k8 = (q & 15) << 3;
        *(uint4*)&lW[r * LSTR + k8] = *(const uint4*)&W1t[r * D + k8];
    }
    __syncthreads();

    const int lane = t & 31, wv = t >> 5;
    const int lmod = lane & 15, lhi = lane >> 4;
    const int m0  = (wv >> 1) * 16;       // 0,16,32,48
    const int n0b = (wv & 1) * 64;        // 0 or 64

    v8f acc[4] = {};
    #pragma unroll
    for (int k0 = 0; k0 < D; k0 += 32) {
        Frag a, b;
        const int kb = k0 + lhi * 8;
        a.q[0] = *(const uint4*)&lA[(m0 + lmod) * LSTR + kb];
        a.q[1] = *(const uint4*)&lA[(m0 + lmod) * LSTR + kb + 16];
        #pragma unroll
        for (int tt = 0; tt < 4; ++tt) {
            int n0 = n0b + tt * 16;
            b.q[0] = *(const uint4*)&lW[(n0 + lmod) * LSTR + kb];
            b.q[1] = *(const uint4*)&lW[(n0 + lmod) * LSTR + kb + 16];
            acc[tt] = __builtin_amdgcn_wmma_f32_16x16x32_bf16(false, a.v, false, b.v,
                                                              (short)0, acc[tt], false, false);
        }
    }
    __syncthreads();                       // all GEMM1 reads of lA/lW done

    // bias + relu -> T (bf16) back into lA ; reload W2t into lW
    #pragma unroll
    for (int tt = 0; tt < 4; ++tt) {
        int n = n0b + tt * 16 + lmod;
        float bias = b1[n];
        #pragma unroll
        for (int j = 0; j < 8; ++j) {
            int m = m0 + j + lhi * 8;
            float v = acc[tt][j] + bias;
            lA[m * LSTR + n] = f2bf(v > 0.f ? v : 0.f);
        }
    }
    #pragma unroll
    for (int c = 0; c < 8; ++c) {
        int q = t + c * 256, r = q >> 4, k8 = (q & 15) << 3;
        *(uint4*)&lW[r * LSTR + k8] = *(const uint4*)&W2t[r * D + k8];
    }
    __syncthreads();

    v8f acc2[4] = {};
    #pragma unroll
    for (int k0 = 0; k0 < D; k0 += 32) {
        Frag a, b;
        const int kb = k0 + lhi * 8;
        a.q[0] = *(const uint4*)&lA[(m0 + lmod) * LSTR + kb];
        a.q[1] = *(const uint4*)&lA[(m0 + lmod) * LSTR + kb + 16];
        #pragma unroll
        for (int tt = 0; tt < 4; ++tt) {
            int n0 = n0b + tt * 16;
            b.q[0] = *(const uint4*)&lW[(n0 + lmod) * LSTR + kb];
            b.q[1] = *(const uint4*)&lW[(n0 + lmod) * LSTR + kb + 16];
            acc2[tt] = __builtin_amdgcn_wmma_f32_16x16x32_bf16(false, a.v, false, b.v,
                                                               (short)0, acc2[tt], false, false);
        }
    }
    #pragma unroll
    for (int tt = 0; tt < 4; ++tt) {
        int n = n0b + tt * 16 + lmod;
        float bias = b2[n];
        #pragma unroll
        for (int j = 0; j < 8; ++j) {
            int m = m0 + j + lhi * 8;
            out[(size_t)(rowBase + m) * D + n] = acc2[tt][j] + bias;
        }
    }
}

// ---------------- score net: s = relu(A@W1+b1) @ w2 + b2  (w2 is Dx1) ----------------
__global__ __launch_bounds__(256)
void score_kernel(const unsigned short* __restrict__ Abf,
                  const unsigned short* __restrict__ W1t,
                  const float* __restrict__ b1,
                  const float* __restrict__ w2,     // [D]
                  const float* __restrict__ b2,     // [1]
                  float* __restrict__ scoresOut)    // [NPAD]
{
    __shared__ unsigned short lA[TILE_M * LSTR];
    __shared__ unsigned short lW[D * LSTR];
    __shared__ float red[TILE_M];

    const int t = threadIdx.x;
    const int rowBase = blockIdx.x * TILE_M;
    if (t < TILE_M) red[t] = 0.f;

    #pragma unroll
    for (int c = 0; c < 4; ++c) {
        int q = t + c * 256, r = q >> 4, k8 = (q & 15) << 3;
        *(uint4*)&lA[r * LSTR + k8] = *(const uint4*)&Abf[(size_t)(rowBase + r) * D + k8];
    }
    #pragma unroll
    for (int c = 0; c < 8; ++c) {
        int q = t + c * 256, r = q >> 4, k8 = (q & 15) << 3;
        *(uint4*)&lW[r * LSTR + k8] = *(const uint4*)&W1t[r * D + k8];
    }
    __syncthreads();

    const int lane = t & 31, wv = t >> 5;
    const int lmod = lane & 15, lhi = lane >> 4;
    const int m0  = (wv >> 1) * 16;
    const int n0b = (wv & 1) * 64;

    v8f acc[4] = {};
    #pragma unroll
    for (int k0 = 0; k0 < D; k0 += 32) {
        Frag a, b;
        const int kb = k0 + lhi * 8;
        a.q[0] = *(const uint4*)&lA[(m0 + lmod) * LSTR + kb];
        a.q[1] = *(const uint4*)&lA[(m0 + lmod) * LSTR + kb + 16];
        #pragma unroll
        for (int tt = 0; tt < 4; ++tt) {
            int n0 = n0b + tt * 16;
            b.q[0] = *(const uint4*)&lW[(n0 + lmod) * LSTR + kb];
            b.q[1] = *(const uint4*)&lW[(n0 + lmod) * LSTR + kb + 16];
            acc[tt] = __builtin_amdgcn_wmma_f32_16x16x32_bf16(false, a.v, false, b.v,
                                                              (short)0, acc[tt], false, false);
        }
    }

    float p[8] = {0.f, 0.f, 0.f, 0.f, 0.f, 0.f, 0.f, 0.f};
    #pragma unroll
    for (int tt = 0; tt < 4; ++tt) {
        int n = n0b + tt * 16 + lmod;
        float bias = b1[n], wn = w2[n];
        #pragma unroll
        for (int j = 0; j < 8; ++j) {
            float v = acc[tt][j] + bias;
            p[j] += (v > 0.f ? v : 0.f) * wn;
        }
    }
    #pragma unroll
    for (int j = 0; j < 8; ++j) atomicAdd(&red[m0 + j + lhi * 8], p[j]);
    __syncthreads();
    if (t < TILE_M) scoresOut[rowBase + t] = red[t] + b2[0];
}

// ---------------- edge scatter: aggr[dst] += M[src] * w  (1 wave per edge) ----------------
__global__ __launch_bounds__(256)
void scatter_kernel(const float* __restrict__ M, const float* __restrict__ ew,
                    const int* __restrict__ src, const int* __restrict__ dst,
                    float* __restrict__ aggr) {
    int gw = (blockIdx.x * 256 + threadIdx.x) >> 5;
    int lane = threadIdx.x & 31;
    if (gw >= NE) return;
    int s = src[gw], d = dst[gw];
    float w = ew[gw];
    #pragma unroll
    for (int k = 0; k < 4; ++k) {
        int c = lane + k * 32;
        float v = M[(size_t)s * D + c] * w;
        unsafeAtomicAdd(&aggr[(size_t)d * D + c], v);
    }
}

// ---------------- h = relu(aggr) -> stacked[l] (f32) and hbf (bf16) ----------------
__global__ __launch_bounds__(256)
void relu_out_kernel(const float* __restrict__ aggr, float* __restrict__ st,
                     unsigned short* __restrict__ hbf) {
    int i = blockIdx.x * 256 + threadIdx.x;
    if (i < NN * D) {
        float v = aggr[i];
        v = v > 0.f ? v : 0.f;
        st[i] = v;
        hbf[i] = f2bf(v);
    }
}

// ---------------- softmax over layers + weighted sum ----------------
__global__ __launch_bounds__(256)
void finalize_kernel(const float* __restrict__ scores, float* __restrict__ dout) {
    int n = blockIdx.x * 256 + threadIdx.x;
    if (n >= NN) return;
    float s0 = scores[n], s1 = scores[NPAD + n], s2 = scores[2 * NPAD + n];
    float mx = fmaxf(s0, fmaxf(s1, s2));
    float e0 = __expf(s0 - mx), e1 = __expf(s1 - mx), e2 = __expf(s2 - mx);
    float inv = 1.f / (e0 + e1 + e2);
    float w0 = e0 * inv, w1 = e1 * inv, w2 = e2 * inv;
    float* lw = dout + (size_t)4 * NN * D;
    lw[n] = w0; lw[NN + n] = w1; lw[2 * NN + n] = w2;
    const float* st0 = dout + (size_t)NN * D + (size_t)n * D;
    const float* st1 = st0 + (size_t)NN * D;
    const float* st2 = st1 + (size_t)NN * D;
    float* o = dout + (size_t)n * D;
    #pragma unroll
    for (int d = 0; d < D; d += 4) {
        float4 a = *(const float4*)&st0[d];
        float4 b = *(const float4*)&st1[d];
        float4 c = *(const float4*)&st2[d];
        float4 r;
        r.x = w0 * a.x + w1 * b.x + w2 * c.x;
        r.y = w0 * a.y + w1 * b.y + w2 * c.y;
        r.z = w0 * a.z + w1 * b.z + w2 * c.z;
        r.w = w0 * a.w + w1 * b.w + w2 * c.w;
        *(float4*)&o[d] = r;
    }
}

extern "C" void kernel_launch(void* const* d_in, const int* in_sizes, int n_in,
                              void* d_out, int out_size, void* d_ws, size_t ws_size,
                              hipStream_t stream) {
    const float* x      = (const float*)d_in[0];
    const int*   ei     = (const int*)d_in[1];
    const float* conf   = (const float*)d_in[2];
    const float* msgW1  = (const float*)d_in[3];
    const float* msgb1  = (const float*)d_in[4];
    const float* msgW2  = (const float*)d_in[5];
    const float* msgb2  = (const float*)d_in[6];
    const float* selfW1 = (const float*)d_in[7];
    const float* selfb1 = (const float*)d_in[8];
    const float* selfW2 = (const float*)d_in[9];
    const float* selfb2 = (const float*)d_in[10];
    const float* scW1   = (const float*)d_in[11];
    const float* scb1   = (const float*)d_in[12];
    const float* scW2   = (const float*)d_in[13];
    const float* scb2   = (const float*)d_in[14];

    const int* src = ei;
    const int* dst = ei + NE;

    float* out = (float*)d_out;
    float* stacked = out + (size_t)NN * D;   // [3][NN][D]

    char* ws = (char*)d_ws;
    size_t off = 0;
    auto alloc = [&](size_t bytes) -> char* {
        char* p = ws + off;
        off += (bytes + 255) & ~(size_t)255;
        return p;
    };
    float*          ew     = (float*)alloc((size_t)NE * 4);
    unsigned short* hbf    = (unsigned short*)alloc((size_t)NPAD * D * 2);
    float*          M      = (float*)alloc((size_t)NPAD * D * 4);
    float*          aggr   = (float*)alloc((size_t)NPAD * D * 4);
    float*          scores = (float*)alloc((size_t)3 * NPAD * 4);
    unsigned short* wts    = (unsigned short*)alloc((size_t)13 * D * D * 2);
    (void)ws_size; (void)in_sizes; (void)n_in; (void)out_size;

    edgew_kernel<<<(NE + 255) / 256, 256, 0, stream>>>(src, dst, conf, ew);
    cvt_kernel<<<(NN * D) / 256, 256, 0, stream>>>(x, hbf, NN * D);

    for (int i = 0; i < 3; ++i) {
        wcvt_kernel<<<64, 256, 0, stream>>>(msgW1  + (size_t)i * D * D, wts + (size_t)(0 + i) * D * D);
        wcvt_kernel<<<64, 256, 0, stream>>>(msgW2  + (size_t)i * D * D, wts + (size_t)(3 + i) * D * D);
        wcvt_kernel<<<64, 256, 0, stream>>>(selfW1 + (size_t)i * D * D, wts + (size_t)(6 + i) * D * D);
        wcvt_kernel<<<64, 256, 0, stream>>>(selfW2 + (size_t)i * D * D, wts + (size_t)(9 + i) * D * D);
    }
    wcvt_kernel<<<64, 256, 0, stream>>>(scW1, wts + (size_t)12 * D * D);

    const int gB = NPAD / TILE_M;   // 158
    for (int i = 0; i < 3; ++i) {
        // self MLP -> aggr (scatter accumulates on top; no zero-init pass)
        mlp2_kernel<<<gB, 256, 0, stream>>>(hbf, wts + (size_t)(6 + i) * D * D, selfb1 + i * D,
                                            wts + (size_t)(9 + i) * D * D, selfb2 + i * D, aggr);
        // message MLP (per-node, not per-edge) -> M
        mlp2_kernel<<<gB, 256, 0, stream>>>(hbf, wts + (size_t)(0 + i) * D * D, msgb1 + i * D,
                                            wts + (size_t)(3 + i) * D * D, msgb2 + i * D, M);
        scatter_kernel<<<NE / 8, 256, 0, stream>>>(M, ew, src, dst, aggr);
        relu_out_kernel<<<(NN * D) / 256, 256, 0, stream>>>(aggr, stacked + (size_t)i * NN * D, hbf);
        score_kernel<<<gB, 256, 0, stream>>>(hbf, wts + (size_t)12 * D * D, scb1, scW2, scb2,
                                             scores + (size_t)i * NPAD);
    }
    finalize_kernel<<<(NN + 255) / 256, 256, 0, stream>>>(scores, out);
}
